// StepwiseFrameLSTM_68401649156384
// MI455X (gfx1250) — compile-verified, matrix operand order checked by reading, not run on hardware
//
#include <hip/hip_runtime.h>
#include <hip/hip_bf16.h>
#include <math.h>

// ---------------- problem constants ----------------
#define T_STEPS   256
#define IN_SIZE   12288           // K of GEMM1
#define HID       1024
#define GATES     4096            // 4*HID, N of GEMM1
#define NB_LSTM   32              // persistent blocks for recurrence

typedef __attribute__((ext_vector_type(2)))  float           v2f;
typedef __attribute__((ext_vector_type(8)))  float           v8f;
typedef __attribute__((ext_vector_type(16))) __bf16          v16bf;
typedef __attribute__((ext_vector_type(16))) unsigned short  v16us;

#if __has_builtin(__builtin_amdgcn_wmma_f32_16x16x4_f32)
// =====================================================================
// Kernel 1 (preferred): x_proj = X @ W_ih^T + b  via native f32 WMMA.
// V_WMMA_F32_16X16X4_F32: A 16x4 f32 (2 VGPR/lane), B 4x16 f32, C/D 16x16 f32.
// A layout: lane L holds row (L&15); K = kk + ((L<16)?{0,1}:{2,3}).
// B mirrors A (lane holds W row = B column, same K mapping).
// No precision loss, zero conversion VALU: inner loop is loads + wmma only.
// Grid: 256 blocks (one 16-row N-tile of W_ih each), 256 thr = 8 waves,
// each wave owns M-tiles {w, w+8}; B fragment shared by both.
// =====================================================================
__global__ __launch_bounds__(256) void xproj_wmma_kernel(
    const float* __restrict__ X,      // 256 x 12288
    const float* __restrict__ W,      // 4096 x 12288
    const float* __restrict__ b_ih,   // 4096
    const float* __restrict__ b_hh,   // 4096
    float* __restrict__ xproj)        // 256 x 4096
{
    const int lane  = threadIdx.x & 31;
    const int wave  = threadIdx.x >> 5;
    const int ncol0 = blockIdx.x * 16;
    const int lrow  = lane & 15;
    const int kq    = (lane < 16) ? 0 : 2;   // K-pair selector per half-wave

    const float* wrow  = W + (size_t)(ncol0 + lrow) * IN_SIZE + kq;
    const float* xrow0 = X + (size_t)(wave * 16 + lrow) * IN_SIZE + kq;       // mt = wave
    const float* xrow1 = X + (size_t)((wave + 8) * 16 + lrow) * IN_SIZE + kq; // mt = wave+8

    v8f acc0 = {};
    v8f acc1 = {};

#pragma unroll 8
    for (int kk = 0; kk < IN_SIZE; kk += 4) {
        v2f b  = *reinterpret_cast<const v2f*>(wrow + kk);
        v2f a0 = *reinterpret_cast<const v2f*>(xrow0 + kk);
        v2f a1 = *reinterpret_cast<const v2f*>(xrow1 + kk);
        acc0 = __builtin_amdgcn_wmma_f32_16x16x4_f32(false, a0, false, b, (short)0, acc0, false, false);
        acc1 = __builtin_amdgcn_wmma_f32_16x16x4_f32(false, a1, false, b, (short)0, acc1, false, false);
    }

    const int n    = ncol0 + (lane & 15);
    const float bb = b_ih[n] + b_hh[n];
    const int moff = (lane < 16) ? 0 : 8;
#pragma unroll
    for (int r = 0; r < 8; ++r) {
        int m0 = wave * 16 + r + moff;
        int m1 = (wave + 8) * 16 + r + moff;
        xproj[(size_t)m0 * GATES + n] = acc0[r] + bb;
        xproj[(size_t)m1 * GATES + n] = acc1[r] + bb;
    }
}

#else  // ----- fallback: bf16x2 split WMMA (known-good path) -----

__device__ __forceinline__ unsigned short f32_to_bf16(float f) {
    unsigned u = __float_as_uint(f);
    unsigned r = u + 0x7FFFu + ((u >> 16) & 1u);
    return (unsigned short)(r >> 16);
}
__device__ __forceinline__ float bf16_to_f32(unsigned short h) {
    return __uint_as_float(((unsigned)h) << 16);
}

__device__ __forceinline__ void make_frags(const float* __restrict__ rowp, int kbase,
                                           v16bf& hi, v16bf& lo) {
    const float4* p0 = reinterpret_cast<const float4*>(rowp + kbase);
    const float4* p1 = reinterpret_cast<const float4*>(rowp + kbase + 16);
    float4 q0 = p0[0], q1 = p0[1], q2 = p1[0], q3 = p1[1];
    float f[16] = {q0.x,q0.y,q0.z,q0.w, q1.x,q1.y,q1.z,q1.w,
                   q2.x,q2.y,q2.z,q2.w, q3.x,q3.y,q3.z,q3.w};
    v16us h, l;
#pragma unroll
    for (int e = 0; e < 16; ++e) {
        unsigned short hb = f32_to_bf16(f[e]);
        h[e] = hb;
        l[e] = f32_to_bf16(f[e] - bf16_to_f32(hb));
    }
    hi = __builtin_bit_cast(v16bf, h);
    lo = __builtin_bit_cast(v16bf, l);
}

__global__ __launch_bounds__(256) void xproj_wmma_kernel(
    const float* __restrict__ X, const float* __restrict__ W,
    const float* __restrict__ b_ih, const float* __restrict__ b_hh,
    float* __restrict__ xproj)
{
    const int lane  = threadIdx.x & 31;
    const int wave  = threadIdx.x >> 5;
    const int ncol0 = blockIdx.x * 16;
    const int lrow  = lane & 15;
    const int khalf = (lane < 16) ? 0 : 8;

    const float* wrow  = W + (size_t)(ncol0 + lrow) * IN_SIZE;
    const float* xrow0 = X + (size_t)(wave * 16 + lrow) * IN_SIZE;
    const float* xrow1 = X + (size_t)((wave + 8) * 16 + lrow) * IN_SIZE;

    v8f acc0 = {};
    v8f acc1 = {};
    for (int kk = 0; kk < IN_SIZE; kk += 32) {
        v16bf bhi, blo, ahi, alo;
        make_frags(wrow, kk + khalf, bhi, blo);
        make_frags(xrow0, kk + khalf, ahi, alo);
        acc0 = __builtin_amdgcn_wmma_f32_16x16x32_bf16(false, alo, false, bhi, (short)0, acc0, false, false);
        acc0 = __builtin_amdgcn_wmma_f32_16x16x32_bf16(false, ahi, false, blo, (short)0, acc0, false, false);
        acc0 = __builtin_amdgcn_wmma_f32_16x16x32_bf16(false, ahi, false, bhi, (short)0, acc0, false, false);
        make_frags(xrow1, kk + khalf, ahi, alo);
        acc1 = __builtin_amdgcn_wmma_f32_16x16x32_bf16(false, alo, false, bhi, (short)0, acc1, false, false);
        acc1 = __builtin_amdgcn_wmma_f32_16x16x32_bf16(false, ahi, false, blo, (short)0, acc1, false, false);
        acc1 = __builtin_amdgcn_wmma_f32_16x16x32_bf16(false, ahi, false, bhi, (short)0, acc1, false, false);
    }

    const int n    = ncol0 + (lane & 15);
    const float bb = b_ih[n] + b_hh[n];
    const int moff = (lane < 16) ? 0 : 8;
#pragma unroll
    for (int r = 0; r < 8; ++r) {
        int m0 = wave * 16 + r + moff;
        int m1 = (wave + 8) * 16 + r + moff;
        xproj[(size_t)m0 * GATES + n] = acc0[r] + bb;
        xproj[(size_t)m1 * GATES + n] = acc1[r] + bb;
    }
}
#endif

// ---------------- Kernel 2: sequential LSTM recurrence (persistent, grid-synced) ----------------
__device__ __forceinline__ float sigmoidf_(float x) { return 1.0f / (1.0f + __expf(-x)); }

__global__ __launch_bounds__(256) void lstm_seq_kernel(
    const float* __restrict__ xproj,  // 256 x 4096
    const float* __restrict__ W_hh,   // 4096 x 1024
    float* __restrict__ hbuf,         // 2 x 1024 (double buffer)
    unsigned* __restrict__ cnt)       // grid barrier counter
{
    __shared__ float h_s[HID];
    __shared__ float gate_s[128];
    __shared__ float c_s[32];

    const int tid  = threadIdx.x;
    const int lane = tid & 31;
    const int wave = tid >> 5;
    const int blk  = blockIdx.x;      // 0..NB_LSTM-1
    const int j0   = blk * 32;        // h slice owned by this block

    if (tid < 32) c_s[tid] = 0.0f;

    for (int t = 0; t < T_STEPS; ++t) {
        const float* hp = hbuf + (size_t)(t & 1) * HID;
        __syncthreads();
        for (int k = tid; k < HID; k += 256) h_s[k] = hp[k];
        __syncthreads();

        // 8 waves x 16 rows = 128 gate rows (4 gates x 32 h-elems)
        for (int rr = 0; rr < 16; ++rr) {
            int rloc = wave * 16 + rr;         // 0..127
            int g    = rloc >> 5;              // gate index
            int jj   = rloc & 31;              // h index within slice
            const float* wr = W_hh + (size_t)(g * HID + j0 + jj) * HID;
            float s = 0.0f;
            for (int k = lane; k < HID; k += 32) s += wr[k] * h_s[k];
#pragma unroll
            for (int off = 16; off; off >>= 1) s += __shfl_xor(s, off, 32);
            if (lane == 0)
                gate_s[rloc] = s + xproj[(size_t)t * GATES + g * HID + j0 + jj];
        }
        __syncthreads();

        if (tid < 32) {
            float gi = gate_s[0 * 32 + tid];
            float gf = gate_s[1 * 32 + tid];
            float gg = gate_s[2 * 32 + tid];
            float go = gate_s[3 * 32 + tid];
            float i_ = sigmoidf_(gi);
            float f_ = sigmoidf_(gf);
            float g_ = tanhf(gg);
            float o_ = sigmoidf_(go);
            float c  = f_ * c_s[tid] + i_ * g_;
            c_s[tid] = c;
            hbuf[(size_t)((t + 1) & 1) * HID + j0 + tid] = o_ * tanhf(c);
        }
        __syncthreads();

        // device-scope release/acquire grid barrier
        if (tid == 0) {
            __hip_atomic_fetch_add(cnt, 1u, __ATOMIC_ACQ_REL, __HIP_MEMORY_SCOPE_AGENT);
            unsigned tgt = (unsigned)NB_LSTM * (unsigned)(t + 1);
            while (__hip_atomic_load(cnt, __ATOMIC_ACQUIRE, __HIP_MEMORY_SCOPE_AGENT) < tgt)
                __builtin_amdgcn_s_sleep(2);
        }
        __syncthreads();
    }
}

// ---------------- Kernel 3: out = h_T @ fc_w^T + fc_b (wave-per-row GEMV) ----------------
__global__ __launch_bounds__(256) void fc_out_kernel(
    const float* __restrict__ hT,     // 1024 (hbuf[0] after 256 steps)
    const float* __restrict__ fc_w,   // 12288 x 1024
    const float* __restrict__ fc_b,   // 12288
    float* __restrict__ out)          // 12288
{
    const int row  = blockIdx.x * 8 + (threadIdx.x >> 5);
    const int lane = threadIdx.x & 31;
    if (row >= IN_SIZE) return;
    const float* wr = fc_w + (size_t)row * HID;
    float s = 0.0f;
    for (int k = lane; k < HID; k += 32) s += wr[k] * hT[k];
#pragma unroll
    for (int off = 16; off; off >>= 1) s += __shfl_xor(s, off, 32);
    if (lane == 0) out[row] = s + fc_b[row];
}

// ---------------- init: zero h double-buffer + barrier counter ----------------
__global__ void init_ws_kernel(float* __restrict__ hbuf, unsigned* __restrict__ cnt) {
    int i = blockIdx.x * 256 + threadIdx.x;
    if (i < 2 * HID) hbuf[i] = 0.0f;
    if (i == 0) *cnt = 0u;
}

extern "C" void kernel_launch(void* const* d_in, const int* in_sizes, int n_in,
                              void* d_out, int out_size, void* d_ws, size_t ws_size,
                              hipStream_t stream) {
    (void)in_sizes; (void)n_in; (void)out_size; (void)ws_size;
    const float* frames = (const float*)d_in[0];  // 256*3*64*64
    const float* W_ih   = (const float*)d_in[1];  // 4096 x 12288
    const float* W_hh   = (const float*)d_in[2];  // 4096 x 1024
    const float* b_ih   = (const float*)d_in[3];  // 4096
    const float* b_hh   = (const float*)d_in[4];  // 4096
    const float* fc_w   = (const float*)d_in[5];  // 12288 x 1024
    const float* fc_b   = (const float*)d_in[6];  // 12288
    float* out = (float*)d_out;

    float*    ws    = (float*)d_ws;
    float*    xproj = ws;                                  // 256*4096 f32
    float*    hbuf  = ws + (size_t)T_STEPS * GATES;        // 2*1024 f32
    unsigned* cnt   = (unsigned*)(hbuf + 2 * HID);         // 1 u32

    init_ws_kernel<<<8, 256, 0, stream>>>(hbuf, cnt);
    xproj_wmma_kernel<<<GATES / 16, 256, 0, stream>>>(frames, W_ih, b_ih, b_hh, xproj);
    lstm_seq_kernel<<<NB_LSTM, 256, 0, stream>>>(xproj, W_hh, hbuf, cnt);
    // after 256 steps, h_T lives in hbuf[(256)&1] == hbuf[0]
    fc_out_kernel<<<IN_SIZE / 8, 256, 0, stream>>>(hbuf, fc_w, fc_b, out);
}
